// Attention_31507880084087
// MI455X (gfx1250) — matrix operand order repaired and checked by reading, not compiled, not run on hardware
//
#include <hip/hip_runtime.h>
#include <hip/hip_bf16.h>

// ---------------------------------------------------------------------------
// Attention layer for MI455X (gfx1250, wave32, WMMA bf16).
//   B=2 S=2048 HID=2048 H=16 HKV=8 D=128  (GQA group = 2)
// Pipeline:
//   cvt_transpose: weights fp32 -> bf16, pre-transposed to k-major (one pass)
//   gemm_qkv_rope: 64x128 tiles, async double-buffered B, reg-prefetched A,
//                  fused RoPE epilogue -> Q[b,h,s,d], K^T[b,h,d,s], V[b,h,s,d]
//   flash_attn:    async double-buffered K/V chunks in LDS, online softmax
//   gemm_out:      64x128 tiles, fully async staging, f32 output
// ---------------------------------------------------------------------------

typedef __attribute__((ext_vector_type(16))) __bf16 v16bf;
typedef __attribute__((ext_vector_type(8)))  float  v8f;

#define B_    2
#define S_    2048
#define HID_  2048
#define H_    16
#define HKV_  8
#define D_    128
#define NO_   4096            // (H + 2*HKV) * D
#define SCALE_ 0.08838834764831845f   // 1/sqrt(128)

union Frag16 { v16bf v; uint4 u[2]; };

__device__ __forceinline__ v8f wmma_bf16(const Frag16& a, const Frag16& b, v8f c) {
  return __builtin_amdgcn_wmma_f32_16x16x32_bf16(false, a.v, false, b.v,
                                                 (short)0, c, false, false);
}

#define PA_ 40    // A tile pitch  [m][k] 64x32 bf16
#define PB_ 136   // B tile pitch  [k][n] 32x128 bf16
#define PC_ 129   // epilogue f32 pitch

// ---------------------------------------------------------------------------
// Weight cast + transpose: src [N][K] f32 -> dst [K][N] bf16 (32x32 LDS tile).
// Makes GEMM B-staging a contiguous copy (async-to-LDS friendly, no VALU).
// ---------------------------------------------------------------------------
__global__ __launch_bounds__(128) void cvt_transpose(
    const float* __restrict__ src, __bf16* __restrict__ dst, int N, int K) {
  __shared__ __bf16 tile[32 * PA_];          // [n][k]
  const int k0 = blockIdx.x * 32, n0 = blockIdx.y * 32;
  const int t = threadIdx.x;
  {
    int n = t >> 2, j0 = (t & 3) * 8;
    const float* s = src + (size_t)(n0 + n) * K + k0 + j0;
    float4 f0 = ((const float4*)s)[0];
    float4 f1 = ((const float4*)s)[1];
    __bf16* d = tile + n * PA_ + j0;
    d[0] = (__bf16)f0.x; d[1] = (__bf16)f0.y; d[2] = (__bf16)f0.z; d[3] = (__bf16)f0.w;
    d[4] = (__bf16)f1.x; d[5] = (__bf16)f1.y; d[6] = (__bf16)f1.z; d[7] = (__bf16)f1.w;
  }
  __syncthreads();
  {
    int k = t >> 2, i0 = (t & 3) * 8;
    union { __bf16 h[8]; uint4 u; } o;
    #pragma unroll
    for (int e = 0; e < 8; ++e) o.h[e] = tile[(i0 + e) * PA_ + k];
    *(uint4*)(dst + (size_t)(k0 + k) * N + n0 + i0) = o.u;
  }
}

// ---------------------------------------------------------------------------
// GEMM1: qkv[row, o] = sum_k x[row,k] * wqT[k, o]
// 64(M) x 128(N) tile, 4 waves x (4 M-blocks x 2 N-blocks) = 8 WMMA / k-step.
// B: async double-buffered copy from pre-transposed weights.
// A: fp32 prefetched to regs, converted to bf16 into double-buffered LDS.
// Epilogue: RoPE (q/k) fused; writes Q, K^T, V bf16.
// ---------------------------------------------------------------------------
#define SMEM1_ (64 * PC_ * 4)   // 33024 B; staging (2*5120 + 2*8704 = 27648) fits

__global__ __launch_bounds__(128) void gemm_qkv_rope(
    const float* __restrict__ x,        // [B*S, HID] f32
    const __bf16* __restrict__ wqT,     // [HID, NO] bf16 (transposed)
    const float* __restrict__ cosT,     // [S, D]
    const float* __restrict__ sinT,     // [S, D]
    __bf16* __restrict__ q16,           // [B, H, S, D]
    __bf16* __restrict__ kT16,          // [B, HKV, D, S]
    __bf16* __restrict__ v16)           // [B, HKV, S, D]
{
  __shared__ __align__(16) char smem[SMEM1_];
  __bf16* sA = (__bf16*)smem;                        // [2][64*PA_]
  __bf16* sB = (__bf16*)(smem + 2 * 64 * PA_ * 2);   // [2][32*PB_]

  const int tid  = threadIdx.x;
  const int lane = tid & 31, wv = tid >> 5;
  const int hi   = lane >> 4, ln = lane & 15;
  const int row0 = blockIdx.x * 64;                  // over B*S
  const int hh   = blockIdx.y;                       // head slice 0..31
  const int o0   = hh * 128;

  // async copy of B k-chunk (32 x 128 bf16, contiguous 256B rows) into LDS
  auto issueB = [&](int kb, int bufi) {
    uint32_t bl = (uint32_t)(size_t)(sB + bufi * (32 * PB_));
    #pragma unroll
    for (int j = 0; j < 4; ++j) {
      int idx = tid + 128 * j;
      int k = idx >> 4, seg = idx & 15;
      unsigned long long ga =
          (unsigned long long)(size_t)(wqT + (size_t)(kb + k) * NO_ + o0 + seg * 8);
      uint32_t la = bl + (uint32_t)(k * PB_ + seg * 8) * 2;
      asm volatile("global_load_async_to_lds_b128 %0, %1, off"
                   :: "v"(la), "v"(ga) : "memory");
    }
  };

  const int am = tid >> 1, ako = (tid & 1) * 16;
  float4 ar[4];
  auto loadA = [&](int kb) {
    const float* src = x + (size_t)(row0 + am) * HID_ + kb + ako;
    ar[0] = ((const float4*)src)[0];
    ar[1] = ((const float4*)src)[1];
    ar[2] = ((const float4*)src)[2];
    ar[3] = ((const float4*)src)[3];
  };
  auto storeA = [&](int bufi) {
    __bf16* d = sA + bufi * (64 * PA_) + am * PA_ + ako;
    d[0]  = (__bf16)ar[0].x; d[1]  = (__bf16)ar[0].y; d[2]  = (__bf16)ar[0].z; d[3]  = (__bf16)ar[0].w;
    d[4]  = (__bf16)ar[1].x; d[5]  = (__bf16)ar[1].y; d[6]  = (__bf16)ar[1].z; d[7]  = (__bf16)ar[1].w;
    d[8]  = (__bf16)ar[2].x; d[9]  = (__bf16)ar[2].y; d[10] = (__bf16)ar[2].z; d[11] = (__bf16)ar[2].w;
    d[12] = (__bf16)ar[3].x; d[13] = (__bf16)ar[3].y; d[14] = (__bf16)ar[3].z; d[15] = (__bf16)ar[3].w;
  };

  v8f acc[4][2];
  for (int a = 0; a < 4; ++a)
    for (int b = 0; b < 2; ++b)
      for (int i = 0; i < 8; ++i) acc[a][b][i] = 0.0f;

  issueB(0, 0);
  loadA(0);
  storeA(0);

  for (int it = 0; it < HID_ / 32; ++it) {
    const int buf = it & 1;
    asm volatile("s_wait_asynccnt 0x0" ::: "memory");
    __syncthreads();
    const bool more = (it + 1) < HID_ / 32;
    if (more) {
      issueB((it + 1) * 32, buf ^ 1);
      loadA((it + 1) * 32);
    }
    // ---- 8 WMMA from LDS buffers ----
    const __bf16* sAb = sA + buf * (64 * PA_);
    const __bf16* sBb = sB + buf * (32 * PB_);
    Frag16 af[4], bfr[2];
    #pragma unroll
    for (int mb = 0; mb < 4; ++mb) {
      const char* base = (const char*)(sAb + (mb * 16 + ln) * PA_);
      af[mb].u[0] = *(const uint4*)(base + 16 * hi);
      af[mb].u[1] = *(const uint4*)(base + 32 + 16 * hi);
    }
    #pragma unroll
    for (int nb = 0; nb < 2; ++nb) {
      const __bf16* base = sBb + wv * 32 + nb * 16 + ln;
      #pragma unroll
      for (int s = 0; s < 16; ++s)
        bfr[nb].v[s] = base[((s & 8) * 2 + 8 * hi + (s & 7)) * PB_];
    }
    #pragma unroll
    for (int mb = 0; mb < 4; ++mb)
      #pragma unroll
      for (int nb = 0; nb < 2; ++nb)
        acc[mb][nb] = wmma_bf16(af[mb], bfr[nb], acc[mb][nb]);
    if (more) storeA(buf ^ 1);
  }
  __syncthreads();

  // ---- epilogue: accum -> LDS f32 [64][PC_], then RoPE + scatter ----
  float* sC = (float*)smem;
  #pragma unroll
  for (int mb = 0; mb < 4; ++mb)
    #pragma unroll
    for (int nb = 0; nb < 2; ++nb)
      #pragma unroll
      for (int i = 0; i < 8; ++i)
        sC[(mb * 16 + i + 8 * hi) * PC_ + wv * 32 + nb * 16 + ln] = acc[mb][nb][i];
  __syncthreads();

  if (hh < 16) {                       // ---- Q head: RoPE + scale, col-fast ----
    int col = tid;
    for (int row = 0; row < 64; ++row) {
      int grow = row0 + row, b = grow >> 11, s = grow & (S_ - 1);
      float v  = sC[row * PC_ + col];
      float pr = sC[row * PC_ + (col ^ 64)];
      float rot = (col < 64) ? -pr : pr;
      float r = (v * cosT[s * D_ + col] + rot * sinT[s * D_ + col]) * SCALE_;
      q16[((size_t)(b * H_ + hh) * S_ + s) * D_ + col] = (__bf16)r;
    }
  } else if (hh < 24) {                // ---- K head: RoPE, row-fast (for kT) ----
    int hk = hh - 16;
    int row = tid & 63, cb = tid >> 6;
    int grow = row0 + row, b = grow >> 11, s = grow & (S_ - 1);
    for (int j = 0; j < 64; ++j) {
      int col = cb * 64 + j;
      float v  = sC[row * PC_ + col];
      float pr = sC[row * PC_ + (col ^ 64)];
      float rot = (col < 64) ? -pr : pr;
      float r = v * cosT[s * D_ + col] + rot * sinT[s * D_ + col];
      kT16[((size_t)(b * HKV_ + hk) * D_ + col) * S_ + s] = (__bf16)r;
    }
  } else {                             // ---- V head: passthrough, col-fast ----
    int hv = hh - 24;
    int col = tid;
    for (int row = 0; row < 64; ++row) {
      int grow = row0 + row, b = grow >> 11, s = grow & (S_ - 1);
      v16[((size_t)(b * HKV_ + hv) * S_ + s) * D_ + col] = (__bf16)sC[row * PC_ + col];
    }
  }
}

// ---------------------------------------------------------------------------
// Flash attention. One wave owns 16 query rows (persistent Q fragments).
// 32-key chunks of K^T ([d][t]) and V ([t][d]) staged cooperatively via
// global_load_async_to_lds_b128, double-buffered; one barrier per chunk:
//   wait(own asyncs) -> barrier (all staged + prev frags consumed) ->
//   issue next chunk -> 16 WMMAs.
// ---------------------------------------------------------------------------
#define PP_  40    // P tile pitch   (16 x 32)
#define SKP_ 40    // K tile pitch   ([d][t] 128 x 32)
#define SVP_ 136   // V tile pitch   ([t][d] 32 x 128)

__global__ __launch_bounds__(128) void flash_attn(
    const __bf16* __restrict__ q16,    // [B, H, S, D]
    const __bf16* __restrict__ kT16,   // [B, HKV, D, S]
    const __bf16* __restrict__ v16,    // [B, HKV, S, D]
    __bf16* __restrict__ attn16)       // [B, S, H*D]
{
  __shared__ __align__(16) __bf16 sK[2][128 * SKP_];
  __shared__ __align__(16) __bf16 sV[2][32 * SVP_];
  __shared__ __align__(16) __bf16 sP[4][16 * PP_];

  const int tid  = threadIdx.x;
  const int lane = tid & 31, wv = tid >> 5;
  const int hi   = lane >> 4, ln = lane & 15;

  const int st = blockIdx.x & 31;               // S/64 query tiles
  const int h  = (blockIdx.x >> 5) & 15;
  const int b  = blockIdx.x >> 9;
  const int kvh = h >> 1;                       // GQA group of 2
  const int s0  = st * 64 + wv * 16;

  const __bf16* qb  = q16 + (size_t)(b * H_ + h) * S_ * D_;
  const __bf16* kbg = kT16 + (size_t)(b * HKV_ + kvh) * D_ * S_;  // [d][S]
  const __bf16* vbg = v16 + (size_t)(b * HKV_ + kvh) * S_ * D_;   // [t][d]
  __bf16* myP = sP[wv];

  auto stage = [&](int t0, int bufi) {
    uint32_t kl = (uint32_t)(size_t)(&sK[bufi][0]);
    uint32_t vl = (uint32_t)(size_t)(&sV[bufi][0]);
    #pragma unroll
    for (int j = 0; j < 4; ++j) {               // K^T chunk: 128 rows x 64B
      int idx = tid + 128 * j;
      int d = idx >> 2, seg = idx & 3;
      unsigned long long ga =
          (unsigned long long)(size_t)(kbg + (size_t)d * S_ + t0 + seg * 8);
      uint32_t la = kl + (uint32_t)(d * SKP_ + seg * 8) * 2;
      asm volatile("global_load_async_to_lds_b128 %0, %1, off"
                   :: "v"(la), "v"(ga) : "memory");
    }
    #pragma unroll
    for (int j = 0; j < 4; ++j) {               // V chunk: 32 rows x 256B
      int idx = tid + 128 * j;
      int t = idx >> 4, seg = idx & 15;
      unsigned long long ga =
          (unsigned long long)(size_t)(vbg + (size_t)(t0 + t) * D_ + seg * 8);
      uint32_t la = vl + (uint32_t)(t * SVP_ + seg * 8) * 2;
      asm volatile("global_load_async_to_lds_b128 %0, %1, off"
                   :: "v"(la), "v"(ga) : "memory");
    }
  };

  // persistent Q fragments (softmax scale folded in at GEMM1 epilogue)
  Frag16 qf[4];
  #pragma unroll
  for (int kk = 0; kk < 4; ++kk) {
    const __bf16* p = qb + (size_t)(s0 + ln) * D_ + kk * 32 + 8 * hi;
    qf[kk].u[0] = *(const uint4*)p;
    qf[kk].u[1] = *(const uint4*)(p + 16);
  }

  float m[8], l[8];
  v8f O[8];
  #pragma unroll
  for (int i = 0; i < 8; ++i) { m[i] = -1e30f; l[i] = 0.0f; }
  #pragma unroll
  for (int f = 0; f < 8; ++f)
    #pragma unroll
    for (int i = 0; i < 8; ++i) O[f][i] = 0.0f;

  stage(0, 0);

  for (int t0 = 0; t0 < S_; t0 += 32) {
    const int buf = (t0 >> 5) & 1;
    asm volatile("s_wait_asynccnt 0x0" ::: "memory");   // own transfers landed
    __syncthreads();                                    // everyone's landed
    if ((t0 + 32) < S_) stage(t0 + 32, buf ^ 1);        // stream next chunk

    // ---- S = Q K^T over D in 4 k-steps x 2 key-halves ----
    v8f sf0, sf1;
    #pragma unroll
    for (int i = 0; i < 8; ++i) { sf0[i] = 0.0f; sf1[i] = 0.0f; }
    #pragma unroll
    for (int kk = 0; kk < 4; ++kk) {
      Frag16 b0, b1;
      const __bf16* kp = &sK[buf][kk * 32 * SKP_ + ln];
      #pragma unroll
      for (int s = 0; s < 16; ++s) {
        int kr = ((s & 8) * 2 + 8 * hi + (s & 7)) * SKP_;
        b0.v[s] = kp[kr];
        b1.v[s] = kp[kr + 16];
      }
      sf0 = wmma_bf16(qf[kk], b0, sf0);
      sf1 = wmma_bf16(qf[kk], b1, sf1);
    }
    // ---- online softmax update (rows r = i + 8*hi) ----
    float alpha[8];
    #pragma unroll
    for (int i = 0; i < 8; ++i) {
      float rm = fmaxf(sf0[i], sf1[i]);
      #pragma unroll
      for (int off = 8; off > 0; off >>= 1)
        rm = fmaxf(rm, __shfl_xor(rm, off, 16));
      float mn = fmaxf(m[i], rm);
      float a  = __expf(m[i] - mn);
      float p0 = __expf(sf0[i] - mn);
      float p1 = __expf(sf1[i] - mn);
      float rs = p0 + p1;
      #pragma unroll
      for (int off = 8; off > 0; off >>= 1)
        rs += __shfl_xor(rs, off, 16);
      l[i] = l[i] * a + rs;
      m[i] = mn;
      alpha[i] = a;
      int r = i + 8 * hi;
      myP[r * PP_ + ln]      = (__bf16)p0;
      myP[r * PP_ + 16 + ln] = (__bf16)p1;
    }
    #pragma unroll
    for (int f = 0; f < 8; ++f)
      #pragma unroll
      for (int i = 0; i < 8; ++i) O[f][i] *= alpha[i];
    // ---- P fragment (A layout) from per-wave LDS bounce (in-order per wave) ----
    Frag16 pf;
    {
      const __bf16* p = myP + ln * PP_ + 8 * hi;
      pf.u[0] = *(const uint4*)p;
      pf.u[1] = *(const uint4*)(p + 16);
    }
    // ---- O += P V over 8 d-blocks ----
    #pragma unroll
    for (int db = 0; db < 8; ++db) {
      Frag16 bv;
      const __bf16* vp = &sV[buf][db * 16 + ln];
      #pragma unroll
      for (int s = 0; s < 16; ++s)
        bv.v[s] = vp[((s & 8) * 2 + 8 * hi + (s & 7)) * SVP_];
      O[db] = wmma_bf16(pf, bv, O[db]);
    }
  }

  // ---- finalize: O / l -> attn16[b, s, h*D + d] (bf16) ----
  #pragma unroll
  for (int i = 0; i < 8; ++i) {
    int srow = s0 + i + 8 * hi;
    float inv = 1.0f / l[i];
    #pragma unroll
    for (int db = 0; db < 8; ++db)
      attn16[((size_t)(b * S_ + srow) * (H_ * D_)) + h * D_ + db * 16 + ln] =
          (__bf16)(O[db][i] * inv);
  }
}

// ---------------------------------------------------------------------------
// GEMM2: out[row, c] = sum_o attn[row, o] * woT[o, c]   (f32 output)
// Same tiling as GEMM1; both A and B staged with async copies (all bf16).
// ---------------------------------------------------------------------------
__global__ __launch_bounds__(128) void gemm_out(
    const __bf16* __restrict__ A,      // [B*S, H*D] bf16
    const __bf16* __restrict__ woT,    // [H*D, HID] bf16 (transposed)
    float* __restrict__ out)           // [B*S, HID] f32
{
  __shared__ __align__(16) __bf16 sA[2][64 * PA_];
  __shared__ __align__(16) __bf16 sB[2][32 * PB_];

  const int tid  = threadIdx.x;
  const int lane = tid & 31, wv = tid >> 5;
  const int hi   = lane >> 4, ln = lane & 15;
  const int row0 = blockIdx.x * 64;
  const int o0   = blockIdx.y * 128;
  const int K    = H_ * D_;

  auto issueA = [&](int kb, int bufi) {
    uint32_t al = (uint32_t)(size_t)(&sA[bufi][0]);
    #pragma unroll
    for (int j = 0; j < 2; ++j) {               // 64 rows x 64B
      int idx = tid + 128 * j;
      int mm = idx >> 2, seg = idx & 3;
      unsigned long long ga =
          (unsigned long long)(size_t)(A + (size_t)(row0 + mm) * K + kb + seg * 8);
      uint32_t la = al + (uint32_t)(mm * PA_ + seg * 8) * 2;
      asm volatile("global_load_async_to_lds_b128 %0, %1, off"
                   :: "v"(la), "v"(ga) : "memory");
    }
  };
  auto issueB = [&](int kb, int bufi) {
    uint32_t bl = (uint32_t)(size_t)(&sB[bufi][0]);
    #pragma unroll
    for (int j = 0; j < 4; ++j) {               // 32 rows x 256B
      int idx = tid + 128 * j;
      int k = idx >> 4, seg = idx & 15;
      unsigned long long ga =
          (unsigned long long)(size_t)(woT + (size_t)(kb + k) * HID_ + o0 + seg * 8);
      uint32_t la = bl + (uint32_t)(k * PB_ + seg * 8) * 2;
      asm volatile("global_load_async_to_lds_b128 %0, %1, off"
                   :: "v"(la), "v"(ga) : "memory");
    }
  };

  v8f acc[4][2];
  for (int a = 0; a < 4; ++a)
    for (int b = 0; b < 2; ++b)
      for (int i = 0; i < 8; ++i) acc[a][b][i] = 0.0f;

  issueA(0, 0);
  issueB(0, 0);

  for (int it = 0; it < K / 32; ++it) {
    const int buf = it & 1;
    asm volatile("s_wait_asynccnt 0x0" ::: "memory");
    __syncthreads();
    if ((it + 1) < K / 32) {
      issueA((it + 1) * 32, buf ^ 1);
      issueB((it + 1) * 32, buf ^ 1);
    }
    Frag16 af[4], bfr[2];
    #pragma unroll
    for (int mb = 0; mb < 4; ++mb) {
      const char* base = (const char*)(&sA[buf][(mb * 16 + ln) * PA_]);
      af[mb].u[0] = *(const uint4*)(base + 16 * hi);
      af[mb].u[1] = *(const uint4*)(base + 32 + 16 * hi);
    }
    #pragma unroll
    for (int nb = 0; nb < 2; ++nb) {
      const __bf16* base = &sB[buf][wv * 32 + nb * 16 + ln];
      #pragma unroll
      for (int s = 0; s < 16; ++s)
        bfr[nb].v[s] = base[((s & 8) * 2 + 8 * hi + (s & 7)) * PB_];
    }
    #pragma unroll
    for (int mb = 0; mb < 4; ++mb)
      #pragma unroll
      for (int nb = 0; nb < 2; ++nb)
        acc[mb][nb] = wmma_bf16(af[mb], bfr[nb], acc[mb][nb]);
  }

  #pragma unroll
  for (int mb = 0; mb < 4; ++mb)
    #pragma unroll
    for (int nb = 0; nb < 2; ++nb) {
      int col = o0 + wv * 32 + nb * 16 + ln;
      #pragma unroll
      for (int i = 0; i < 8; ++i) {
        int grow = row0 + mb * 16 + i + 8 * hi;
        out[(size_t)grow * HID_ + col] = acc[mb][nb][i];
      }
    }
}

// ---------------------------------------------------------------------------
// Host launcher. Workspace layout (bytes, ~72 MB total):
//   wqT16 16MB | woT16 8MB | q16 16MB | kT16 8MB | v16 8MB | attn16 16MB
// ---------------------------------------------------------------------------
extern "C" void kernel_launch(void* const* d_in, const int* in_sizes, int n_in,
                              void* d_out, int out_size, void* d_ws, size_t ws_size,
                              hipStream_t stream) {
  const float* cosT = (const float*)d_in[0];
  const float* sinT = (const float*)d_in[1];
  const float* x    = (const float*)d_in[2];
  const float* wqkv = (const float*)d_in[3];
  const float* wo   = (const float*)d_in[4];
  float* out = (float*)d_out;

  char* ws = (char*)d_ws;
  __bf16* wqT16  = (__bf16*)ws;  ws += (size_t)HID_ * NO_ * 2;
  __bf16* woT16  = (__bf16*)ws;  ws += (size_t)(H_ * D_) * HID_ * 2;
  __bf16* q16    = (__bf16*)ws;  ws += (size_t)B_ * H_ * S_ * D_ * 2;
  __bf16* kT16   = (__bf16*)ws;  ws += (size_t)B_ * HKV_ * D_ * S_ * 2;
  __bf16* v16    = (__bf16*)ws;  ws += (size_t)B_ * HKV_ * S_ * D_ * 2;
  __bf16* attn16 = (__bf16*)ws;

  // w_qkv [NO][HID] -> wqT16 [HID][NO]
  cvt_transpose<<<dim3(HID_ / 32, NO_ / 32), 128, 0, stream>>>(wqkv, wqT16, NO_, HID_);
  // w_o [HID][H*D] -> woT16 [H*D][HID]
  cvt_transpose<<<dim3((H_ * D_) / 32, HID_ / 32), 128, 0, stream>>>(wo, woT16, HID_, H_ * D_);

  gemm_qkv_rope<<<dim3(B_ * S_ / 64, NO_ / 128), 128, 0, stream>>>(
      x, wqT16, cosT, sinT, q16, kT16, v16);

  flash_attn<<<B_ * H_ * (S_ / 64), 128, 0, stream>>>(q16, kT16, v16, attn16);

  gemm_out<<<dim3(B_ * S_ / 64, HID_ / 128), 128, 0, stream>>>(attn16, woT16, out);
}
